// RepPoints_11982958756191
// MI455X (gfx1250) — matrix operand order repaired and checked by reading, not compiled, not run on hardware
//
#include <hip/hip_runtime.h>

// ---------------------------------------------------------------------------
// RepPoints head forward for MI455X (gfx1250):
//  - bf16 NHWC implicit-GEMM 3x3 convs with v_wmma_f32_16x16x32_bf16
//  - input patch staged per-workgroup into LDS via Tensor Data Mover
//    (tensor_load_to_lds + s_wait_tensorcnt), A-fragments via ds_load_b128
//  - fused bilinear-sampled deformable conv feeding WMMA
//  - GroupNorm via LDS tree reduction, epilogues fused into bf16 repack
// ---------------------------------------------------------------------------

typedef __attribute__((ext_vector_type(16))) __bf16 v16bf;
typedef __attribute__((ext_vector_type(8)))  __bf16 v8bf;
typedef __attribute__((ext_vector_type(4)))  __bf16 v4bf;
typedef __attribute__((ext_vector_type(8)))  float  v8f;
typedef __attribute__((ext_vector_type(4)))  unsigned int u32x4;
typedef __attribute__((ext_vector_type(8)))  int  i32x8;
typedef __attribute__((ext_vector_type(4)))  int  i32x4;

#define NB   2
#define CH   256
#define HH   100
#define WW   100
#define PH   102          // padded rows (1 halo each side)
#define PWS  128          // padded row pitch in pixels (halo + TDM slack, zeros)
#define KTOT 2304         // 9 taps * 256 channels
#define GRP  32
#define GEPS 1e-5f

#if defined(__has_builtin)
#if __has_builtin(__builtin_amdgcn_tensor_load_to_lds)
#define HAVE_TDM 1
#endif
#endif
#ifndef HAVE_TDM
#define HAVE_TDM 0
#endif

static __device__ __forceinline__ int imin(int a, int b) { return a < b ? a : b; }
static __device__ __forceinline__ int imax(int a, int b) { return a > b ? a : b; }

static __device__ __forceinline__ void wait_tensorcnt0() {
#if defined(__has_builtin)
#if __has_builtin(__builtin_amdgcn_s_wait_tensorcnt)
    __builtin_amdgcn_s_wait_tensorcnt(0);
    return;
#endif
#endif
    asm volatile("s_wait_tensorcnt 0x0" ::: "memory");
}

// ---------------------------------------------------------------------------
// Weight repack: [O=256][C=256][3][3] fp32 -> bf16 [O][t*256+c]
// ---------------------------------------------------------------------------
__global__ __launch_bounds__(256)
void repack3_kernel(const float* __restrict__ w, __bf16* __restrict__ o2) {
    int tid = blockIdx.x * 256 + threadIdx.x;       // 256*2304 threads
    int r = tid % KTOT;
    int o = tid / KTOT;
    int t = r / CH, c = r % CH;
    o2[tid] = (__bf16)w[((size_t)o * CH + c) * 9 + t];
}

// ---------------------------------------------------------------------------
// Feature fp32 NCHW -> bf16 padded NHWC (ring/slack pre-zeroed by memset)
// ---------------------------------------------------------------------------
__global__ __launch_bounds__(256)
void pad_feat_kernel(const float* __restrict__ in, __bf16* __restrict__ outp) {
    int tid = blockIdx.x * 256 + threadIdx.x;       // 2*256*100*100 threads
    int xx = tid % WW; int t1 = tid / WW;
    int yy = t1 % HH;  int t2 = t1 / HH;
    int c  = t2 % CH;  int n  = t2 / CH;
    outp[((size_t)(n * PH + yy + 1) * PWS + (xx + 1)) * CH + c] = (__bf16)in[tid];
}

// ---------------------------------------------------------------------------
// Stage the 3x18x256 bf16 input patch for a (n, y, x0) workgroup into LDS.
// Uses the Tensor Data Mover (2D descriptor: tile 3 rows x 4608 elems,
// data_size=8B, row stride = PWS*CH*2 bytes). Falls back to a cooperative
// copy if the builtin is unavailable.
// ---------------------------------------------------------------------------
static __device__ __forceinline__ void stage_patch(__bf16* tileA,
                                                   const __bf16* gsrc,
                                                   int wv) {
#if HAVE_TDM
    if (wv == 0) {
        unsigned long long ga = (unsigned long long)(uintptr_t)gsrc;
        unsigned int lds = (unsigned int)(uintptr_t)tileA;   // LDS byte offset
        // D# group 0: count=1, lds_addr, global_addr(57b), type=2
        u32x4 g0 = { 1u, lds, (unsigned int)ga,
                     (unsigned int)(ga >> 32) | (2u << 30) };
        // D# group 1: data_size=3 (8B), tensor dims huge (no OOB zero-fill),
        // tile_dim0 = 18*256*2/8 = 1152, tile_dim1 = 3, stride0 = PWS*CH*2/8
        i32x8 g1;
        g1[0] = 0x00030000;                 // wg_mask=0 | data_size=3<<16
        g1[1] = 0;                          // tensor_dim0[15:0] << 16
        g1[2] = 0x4000;                     // tensor_dim0[31:16] (=2^30 elems)
        g1[3] = 0x4000 | (1152 << 16);      // tensor_dim1 hi | tile_dim0
        g1[4] = 3;                          // tile_dim1=3, tile_dim2=0
        g1[5] = (PWS * CH * 2) / 8;         // tensor_dim0_stride (8B units)
        g1[6] = 0;
        g1[7] = 0;
        i32x4 zz = {0, 0, 0, 0};
#if __clang_major__ >= 23
        i32x8 z8 = {0, 0, 0, 0, 0, 0, 0, 0};
        __builtin_amdgcn_tensor_load_to_lds(g0, g1, zz, zz, z8, 0);
#else
        __builtin_amdgcn_tensor_load_to_lds(g0, g1, zz, zz, 0);
#endif
        wait_tensorcnt0();
    }
    __syncthreads();
#else
    for (int i = threadIdx.x; i < 3 * 18 * (CH / 8); i += 256) {
        int row = i / (18 * (CH / 8));
        int rem = i % (18 * (CH / 8));
        ((v8bf*)tileA)[i] =
            *(const v8bf*)(gsrc + (size_t)row * PWS * CH + (size_t)rem * 8);
    }
    __syncthreads();
#endif
}

// ---------------------------------------------------------------------------
// 3x3 conv as implicit GEMM with WMMA bf16.
//   grid: (7 xtiles, 100 y, 4 = n*2 + nmacro), block 256 (8 waves).
//   A patch staged once into LDS (TDM); each wave: 16 px x 16 outch tile.
//   mode 0: raw fp32 NHWC out (for GroupNorm). mode 1: +bias, ReLU.
// ---------------------------------------------------------------------------
__global__ __launch_bounds__(256)
void conv3_wmma_kernel(const __bf16* __restrict__ pin,   // padded NHWC bf16
                       const __bf16* __restrict__ w2,    // [256][2304]
                       const float*  __restrict__ bias,  // [256] or unused
                       float* __restrict__ out,          // [2][100][100][256]
                       int mode) {
    __shared__ __bf16 tileA[3 * 18 * CH];                // 27648 B

    const int lane = threadIdx.x & 31;
    const int wv   = threadIdx.x >> 5;
    const int x0 = blockIdx.x * 16;
    const int y  = blockIdx.y;
    const int n  = blockIdx.z >> 1;
    const int n0 = (blockIdx.z & 1) * 128 + wv * 16;

    const int m  = lane & 15;
    const int hi = lane >> 4;

    // patch origin: padded row y (=orig y-1+halo), padded col x0 (=orig x0-1+halo)
    const __bf16* __restrict__ gsrc =
        pin + ((size_t)(n * PH + y) * PWS + x0) * CH;
    stage_patch(tileA, gsrc, wv);

    const __bf16* __restrict__ brow = w2 + (size_t)(n0 + m) * KTOT;

    v8f acc = {};
    #pragma unroll
    for (int t = 0; t < 9; ++t) {
        const __bf16* __restrict__ atap =
            tileA + ((t / 3) * 18 + (t % 3) + m) * CH;
        const __bf16* __restrict__ btap = brow + t * CH;
        __builtin_prefetch(btap + CH, 0, 3);             // global_prefetch_b8
        #pragma unroll
        for (int cb = 0; cb < CH; cb += 32) {
            v8bf alo = *(const v8bf*)(atap + cb + hi * 8);        // ds_load_b128
            v8bf ahi = *(const v8bf*)(atap + cb + 16 + hi * 8);
            v8bf blo = *(const v8bf*)(btap + cb + hi * 8);
            v8bf bhi = *(const v8bf*)(btap + cb + 16 + hi * 8);
            v16bf a, b;
            #pragma unroll
            for (int i = 0; i < 8; ++i) {
                a[i] = alo[i]; a[i + 8] = ahi[i];
                b[i] = blo[i]; b[i + 8] = bhi[i];
            }
            acc = __builtin_amdgcn_wmma_f32_16x16x32_bf16(
                false, a, false, b, (short)0, acc, false, false);
        }
    }

    const int ch = n0 + (lane & 15);
    const int mbase = 8 * (lane >> 4);
    const float bv = (mode == 1) ? bias[ch] : 0.f;
    #pragma unroll
    for (int r = 0; r < 8; ++r) {
        int px = x0 + mbase + r;                         // C layout: M = r + 8*h
        if (px < WW) {
            float v = acc[r];
            if (mode == 1) v = fmaxf(v + bv, 0.f);
            out[(((size_t)(n * HH) + y) * WW + px) * CH + ch] = v;
        }
    }
}

// ---------------------------------------------------------------------------
// Deformable 3x3 conv: A-fragment built by bilinear sampling of the padded
// bf16 buffer (zero ring == validity masking), then WMMA. ReLU epilogue.
// ---------------------------------------------------------------------------
__global__ __launch_bounds__(256)
void dcn3_wmma_kernel(const __bf16* __restrict__ pin,
                      const float*  __restrict__ offs,  // [2][100][100][18]
                      const __bf16* __restrict__ w2,
                      float* __restrict__ out) {
    const int lane = threadIdx.x & 31;
    const int wv   = threadIdx.x >> 5;
    const int x0 = blockIdx.x * 16;
    const int y  = blockIdx.y;
    const int n  = blockIdx.z >> 1;
    const int n0 = (blockIdx.z & 1) * 128 + wv * 16;

    const int m  = lane & 15;
    const int hi = lane >> 4;
    const int xp = imin(x0 + m, WW - 1);

    const float* __restrict__ po = offs + ((size_t)(n * HH + y) * WW + xp) * 18;
    const __bf16* __restrict__ bimg = pin + (size_t)n * PH * PWS * CH;
    const __bf16* __restrict__ brow = w2 + (size_t)(n0 + m) * KTOT;

    v8f acc = {};
    #pragma unroll
    for (int t = 0; t < 9; ++t) {
        float gy = (float)y  + (float)(t / 3 - 1) + po[2 * t];
        float gx = (float)xp + (float)(t % 3 - 1) + po[2 * t + 1];
        float fy = floorf(gy), fx = floorf(gx);
        float wy = gy - fy,    wx = gx - fx;
        int iy0 = (int)fy, ix0 = (int)fx;
        // clamp to [-1,100]; +1 maps into padded coords; ring holds zeros,
        // exactly reproducing the reference per-corner validity mask.
        int cy0 = imin(imax(iy0,     -1), HH) + 1;
        int cy1 = imin(imax(iy0 + 1, -1), HH) + 1;
        int cx0 = imin(imax(ix0,     -1), WW) + 1;
        int cx1 = imin(imax(ix0 + 1, -1), WW) + 1;
        const __bf16* p00 = bimg + ((size_t)cy0 * PWS + cx0) * CH;
        const __bf16* p01 = bimg + ((size_t)cy0 * PWS + cx1) * CH;
        const __bf16* p10 = bimg + ((size_t)cy1 * PWS + cx0) * CH;
        const __bf16* p11 = bimg + ((size_t)cy1 * PWS + cx1) * CH;
        float w00 = (1.f - wy) * (1.f - wx), w01 = (1.f - wy) * wx;
        float w10 = wy * (1.f - wx),         w11 = wy * wx;
        const __bf16* __restrict__ btap = brow + t * CH;

        #pragma unroll
        for (int cb = 0; cb < CH; cb += 32) {
            const int c0 = cb + hi * 8;
            const int c1 = cb + 16 + hi * 8;
            v8bf q00 = *(const v8bf*)(p00 + c0), q01 = *(const v8bf*)(p01 + c0);
            v8bf q10 = *(const v8bf*)(p10 + c0), q11 = *(const v8bf*)(p11 + c0);
            v8bf r00 = *(const v8bf*)(p00 + c1), r01 = *(const v8bf*)(p01 + c1);
            v8bf r10 = *(const v8bf*)(p10 + c1), r11 = *(const v8bf*)(p11 + c1);
            v8bf blo = *(const v8bf*)(btap + cb + hi * 8);
            v8bf bhi = *(const v8bf*)(btap + cb + 16 + hi * 8);
            v16bf a, b;
            #pragma unroll
            for (int i = 0; i < 8; ++i) {
                float v0 = w00 * (float)q00[i] + w01 * (float)q01[i]
                         + w10 * (float)q10[i] + w11 * (float)q11[i];
                float v1 = w00 * (float)r00[i] + w01 * (float)r01[i]
                         + w10 * (float)r10[i] + w11 * (float)r11[i];
                a[i]     = (__bf16)v0;
                a[i + 8] = (__bf16)v1;
                b[i] = blo[i]; b[i + 8] = bhi[i];
            }
            acc = __builtin_amdgcn_wmma_f32_16x16x32_bf16(
                false, a, false, b, (short)0, acc, false, false);
        }
    }

    const int ch = n0 + (lane & 15);
    const int mbase = 8 * (lane >> 4);
    #pragma unroll
    for (int r = 0; r < 8; ++r) {
        int px = x0 + mbase + r;
        if (px < WW)
            out[(((size_t)(n * HH) + y) * WW + px) * CH + ch] = fmaxf(acc[r], 0.f);
    }
}

// ---------------------------------------------------------------------------
// GroupNorm statistics: one block per (n, group); LDS tree reduction.
// stats[bid*2] = mean, stats[bid*2+1] = rsqrt(var + eps)
// ---------------------------------------------------------------------------
__global__ __launch_bounds__(256)
void gn_stats_kernel(const float* __restrict__ in, float* __restrict__ stats) {
    __shared__ float ssum[256];
    __shared__ float ssq[256];
    const int bid = blockIdx.x;            // n*32 + g
    const int n = bid >> 5, g = bid & 31;
    const float* __restrict__ base = in + (size_t)n * HH * WW * CH + g * 8;
    float s = 0.f, q = 0.f;
    for (int idx = threadIdx.x; idx < HH * WW * 8; idx += 256) {
        int pix = idx >> 3, c = idx & 7;
        float v = base[(size_t)pix * CH + c];
        s += v; q += v * v;
    }
    ssum[threadIdx.x] = s; ssq[threadIdx.x] = q;
    __syncthreads();
    for (int off = 128; off > 0; off >>= 1) {
        if ((int)threadIdx.x < off) {
            ssum[threadIdx.x] += ssum[threadIdx.x + off];
            ssq[threadIdx.x]  += ssq[threadIdx.x + off];
        }
        __syncthreads();
    }
    if (threadIdx.x == 0) {
        const float inv = 1.f / (float)(HH * WW * 8);
        float mean = ssum[0] * inv;
        float var  = ssq[0] * inv - mean * mean;
        stats[bid * 2]     = mean;
        stats[bid * 2 + 1] = rsqrtf(var + GEPS);
    }
}

// ---------------------------------------------------------------------------
// GroupNorm apply + ReLU + bf16 convert into next padded NHWC buffer.
// One thread per 4 channels (never crosses a group of 8).
// ---------------------------------------------------------------------------
__global__ __launch_bounds__(256)
void gn_apply_kernel(const float* __restrict__ in, const float* __restrict__ stats,
                     const float* __restrict__ sc, const float* __restrict__ bi,
                     __bf16* __restrict__ outp) {
    int tid = blockIdx.x * 256 + threadIdx.x;       // 2*1e4*64 threads
    int c4 = (tid % (CH / 4)) * 4;
    int p  = tid / (CH / 4);
    int n = p / (HH * WW); int pix = p % (HH * WW);
    int yy = pix / WW, xx = pix % WW;
    int g = c4 >> 3;
    float mean = stats[(n * GRP + g) * 2];
    float rstd = stats[(n * GRP + g) * 2 + 1];
    const float* __restrict__ xi = in + (size_t)p * CH + c4;
    v4bf o;
    #pragma unroll
    for (int i = 0; i < 4; ++i) {
        float v = (xi[i] - mean) * rstd * sc[c4 + i] + bi[c4 + i];
        o[i] = (__bf16)fmaxf(v, 0.f);
    }
    *(v4bf*)(outp + ((size_t)(n * PH + yy + 1) * PWS + (xx + 1)) * CH + c4) = o;
}

// ---------------------------------------------------------------------------
// 1x1 conv (K=256) with optional bias / residual-add / DCN-offset emission.
//   out NCHW; offsOut (if set, OC==18): pixel-major offsets = pts - base_grid
// ---------------------------------------------------------------------------
__global__ __launch_bounds__(256)
void conv1x1_kernel(const float* __restrict__ in,   // [2*1e4][256] NHWC
                    const float* __restrict__ w,    // [OC][256]
                    const float* __restrict__ b,    // [OC]
                    float* __restrict__ out,        // NCHW
                    int OC,
                    const float* __restrict__ addNCHW,
                    float* __restrict__ offsOut) {
    int tid = blockIdx.x * 256 + threadIdx.x;
    int total = NB * HH * WW * OC;
    if (tid >= total) return;
    int oc = tid % OC, p = tid / OC;
    int n = p / (HH * WW), pix = p % (HH * WW);
    const float* __restrict__ xi = in + (size_t)p * CH;
    const float* __restrict__ wi = w + (size_t)oc * CH;
    float acc = b[oc];
    #pragma unroll 4
    for (int c = 0; c < CH; c += 4)
        acc += xi[c] * wi[c] + xi[c + 1] * wi[c + 1]
             + xi[c + 2] * wi[c + 2] + xi[c + 3] * wi[c + 3];
    size_t oidx = (size_t)(n * OC + oc) * (HH * WW) + pix;
    float res = acc;
    if (addNCHW) res += addNCHW[oidx];
    out[oidx] = res;
    if (offsOut) {
        int k = oc >> 1;
        float base = (oc & 1) ? (float)(k % 3 - 1) : (float)(k / 3 - 1);
        offsOut[(size_t)p * 18 + oc] = acc - base;
    }
}

// ---------------------------------------------------------------------------
// Host-side orchestration
// ---------------------------------------------------------------------------
extern "C" void kernel_launch(void* const* d_in, const int* in_sizes, int n_in,
                              void* d_out, int out_size, void* d_ws, size_t ws_size,
                              hipStream_t stream) {
    const float* feature     = (const float*)d_in[0];
    const float* cls_w       = (const float*)d_in[1];
    const float* cls_gn_s    = (const float*)d_in[2];
    const float* cls_gn_b    = (const float*)d_in[3];
    const float* reg_w       = (const float*)d_in[4];
    const float* reg_gn_s    = (const float*)d_in[5];
    const float* reg_gn_b    = (const float*)d_in[6];
    const float* init_conv_w = (const float*)d_in[7];
    const float* init_conv_b = (const float*)d_in[8];
    const float* init_out_w  = (const float*)d_in[9];
    const float* init_out_b  = (const float*)d_in[10];
    const float* dcn_cls_w   = (const float*)d_in[11];
    const float* cls_out_w   = (const float*)d_in[12];
    const float* cls_out_b   = (const float*)d_in[13];
    const float* dcn_ref_w   = (const float*)d_in[14];
    const float* ref_out_w   = (const float*)d_in[15];
    const float* ref_out_b   = (const float*)d_in[16];

    float* out_cls = (float*)d_out;                       // [2][80][100][100]
    float* out_pts = out_cls + (size_t)NB * 80 * HH * WW; // [2][18][100][100]
    float* out_ref = out_pts + (size_t)NB * 18 * HH * WW; // [2][18][100][100]

    // ---- workspace carve-up --------------------------------------------
    const size_t PAD_BYTES = (size_t)NB * PH * PWS * CH * sizeof(__bf16);
    const size_t PAD_ALGN  = (PAD_BYTES + 511) & ~(size_t)511;
    char* ws = (char*)d_ws;
    size_t off = 0;
    __bf16* P[5];
    for (int i = 0; i < 5; ++i) { P[i] = (__bf16*)(ws + off); off += PAD_ALGN; }
    const size_t pad_region = off;
    float* convtmp = (float*)(ws + off); off += ((size_t)NB * HH * WW * CH * 4 + 511) & ~(size_t)511;
    float* offsbuf = (float*)(ws + off); off += ((size_t)NB * HH * WW * 18 * 4 + 511) & ~(size_t)511;
    float* stats   = (float*)(ws + off); off += 512;
    __bf16* W2[9];
    for (int i = 0; i < 9; ++i) { W2[i] = (__bf16*)(ws + off); off += ((size_t)CH * KTOT * 2 + 511) & ~(size_t)511; }

    // zero the padded buffers (halo ring + TDM slack columns must be zero)
    hipMemsetAsync(ws, 0, pad_region, stream);

    // ---- weight repacks -------------------------------------------------
    const size_t W3SZ = (size_t)CH * CH * 9;
    const float* wsrc[9] = { cls_w, cls_w + W3SZ, cls_w + 2 * W3SZ,
                             reg_w, reg_w + W3SZ, reg_w + 2 * W3SZ,
                             init_conv_w, dcn_cls_w, dcn_ref_w };
    const int rpBlocks = (int)((CH * KTOT) / 256);
    for (int i = 0; i < 9; ++i)
        repack3_kernel<<<rpBlocks, 256, 0, stream>>>(wsrc[i], W2[i]);

    // ---- feature -> bf16 padded NHWC -----------------------------------
    const int elems = NB * CH * HH * WW;
    pad_feat_kernel<<<elems / 256, 256, 0, stream>>>(feature, P[0]);

    const dim3 cg((WW + 15) / 16, HH, NB * 2);   // conv / dcn grid
    const int gnApplyBlocks = (NB * HH * WW * (CH / 4)) / 256;

    // ---- cls / reg towers ----------------------------------------------
    // cls chain: P0 -> P1 -> P2 -> P1   (final cls feat in P1)
    // reg chain: P0 -> P3 -> P4 -> P3   (final reg feat in P3)
    const int cls_chain[4] = {0, 1, 2, 1};
    const int reg_chain[4] = {0, 3, 4, 3};
    for (int i = 0; i < 3; ++i) {
        conv3_wmma_kernel<<<cg, 256, 0, stream>>>(P[cls_chain[i]], W2[i], nullptr, convtmp, 0);
        gn_stats_kernel<<<NB * GRP, 256, 0, stream>>>(convtmp, stats);
        gn_apply_kernel<<<gnApplyBlocks, 256, 0, stream>>>(
            convtmp, stats, cls_gn_s + i * CH, cls_gn_b + i * CH, P[cls_chain[i + 1]]);

        conv3_wmma_kernel<<<cg, 256, 0, stream>>>(P[reg_chain[i]], W2[3 + i], nullptr, convtmp, 0);
        gn_stats_kernel<<<NB * GRP, 256, 0, stream>>>(convtmp, stats);
        gn_apply_kernel<<<gnApplyBlocks, 256, 0, stream>>>(
            convtmp, stats, reg_gn_s + i * CH, reg_gn_b + i * CH, P[reg_chain[i + 1]]);
    }
    __bf16* cls_feat3 = P[1];
    __bf16* reg_feat3 = P[3];

    // ---- points-init branch --------------------------------------------
    conv3_wmma_kernel<<<cg, 256, 0, stream>>>(reg_feat3, W2[6], init_conv_b, convtmp, 1);
    {
        int total = NB * HH * WW * 18;
        conv1x1_kernel<<<(total + 255) / 256, 256, 0, stream>>>(
            convtmp, init_out_w, init_out_b, out_pts, 18, nullptr, offsbuf);
    }

    // ---- deformable cls branch -----------------------------------------
    dcn3_wmma_kernel<<<cg, 256, 0, stream>>>(cls_feat3, offsbuf, W2[7], convtmp);
    {
        int total = NB * HH * WW * 80;
        conv1x1_kernel<<<(total + 255) / 256, 256, 0, stream>>>(
            convtmp, cls_out_w, cls_out_b, out_cls, 80, nullptr, nullptr);
    }

    // ---- deformable refine branch --------------------------------------
    dcn3_wmma_kernel<<<cg, 256, 0, stream>>>(reg_feat3, offsbuf, W2[8], convtmp);
    {
        int total = NB * HH * WW * 18;
        conv1x1_kernel<<<(total + 255) / 256, 256, 0, stream>>>(
            convtmp, ref_out_w, ref_out_b, out_ref, 18, out_pts, nullptr);
    }
}
